// MonoidBlock_12799002542293
// MI455X (gfx1250) — matrix operand order repaired and checked by reading, not compiled, not run on hardware
//
#include <hip/hip_runtime.h>
#include <math.h>

// ---------------------------------------------------------------------------
// Problem constants (from reference)
// ---------------------------------------------------------------------------
#define BSZ    32
#define SEQ    2048
#define DCH    512
#define MBSZ   256
#define NMB    8
#define EXCH   32
#define AMIN   0.9f
#define ARANGE 0.099f      /* A_MAX - A_MIN */
#define INJSC  0.125f      /* 2^-INJ_SHIFT */

typedef __attribute__((ext_vector_type(16))) _Float16 v16h;
typedef __attribute__((ext_vector_type(8)))  float    v8f;

// ---------------------------------------------------------------------------
// Optional precompute: a_tab = AMIN + ARANGE * sigmoid(a_raw), 256x512 table.
// Table is 512 KB -> stays resident in the 192 MB L2 for all later gathers.
// ---------------------------------------------------------------------------
__global__ __launch_bounds__(256)
void atab_kernel(const float* __restrict__ a_raw, float* __restrict__ atab) {
    int i = blockIdx.x * 256 + threadIdx.x;     // 131072 total
    float x = a_raw[i];
    atab[i] = AMIN + ARANGE / (1.0f + expf(-x));
}

// ---------------------------------------------------------------------------
// Fused kernel: one workgroup per batch element runs the entire 8-microbatch
// pipeline.  State h lives in a register for all 2048 steps; tokens in LDS;
// exchange matvec done with v_wmma_f32_16x16x32_f16 (hi/lo f16 split).
// ---------------------------------------------------------------------------
__global__ __launch_bounds__(DCH)
void monoid_fused_kernel(const int*   __restrict__ tokens,
                         const float* __restrict__ a_raw,
                         const float* __restrict__ b_tab,
                         const float* __restrict__ a_tab,   // nullable
                         const float* __restrict__ exw,
                         float*       __restrict__ out) {
    __shared__ float sWn[EXCH * EXCH];   // W / sigma_max
    __shared__ int   toks[SEQ];          // all tokens of this batch element
    __shared__ float sl[DCH];
    __shared__ float su[EXCH], suo[EXCH];
    __shared__ float pv_[EXCH], pt_[EXCH], pr_[EXCH];

    const int tid   = threadIdx.x;
    const int batch = blockIdx.x;
    const int ch    = tid;               // one lane per channel

    for (int i = tid; i < EXCH * EXCH; i += DCH) sWn[i] = exw[i];
    for (int i = tid; i < SEQ; i += DCH) toks[i] = tokens[batch * SEQ + i];
    if (tid < EXCH) pv_[tid] = 1.0f + 0.001f * (float)tid;  // break symmetry
    __syncthreads();

    // ---- spectral norm of W via power iteration (wave 0 works, all sync) --
    for (int it = 0; it < 200; ++it) {
        float ti = 0.f, si = 0.f;
        if (tid < EXCH)                              // t = W v
            for (int k = 0; k < EXCH; ++k) ti += sWn[tid * EXCH + k] * pv_[k];
        __syncthreads();
        if (tid < EXCH) pt_[tid] = ti;
        __syncthreads();
        if (tid < EXCH) {                            // s = W^T t
            for (int j = 0; j < EXCH; ++j) si += sWn[j * EXCH + tid] * pt_[j];
            pr_[tid] = si * si;
        }
        __syncthreads();
        if (tid < EXCH) {
            float s2 = 0.f;
            for (int k = 0; k < EXCH; ++k) s2 += pr_[k];
            pv_[tid] = si * rsqrtf(s2);
        }
        __syncthreads();
    }
    if (tid < EXCH) {                                // sigma = ||W v||
        float ti = 0.f;
        for (int k = 0; k < EXCH; ++k) ti += sWn[tid * EXCH + k] * pv_[k];
        pr_[tid] = ti * ti;
    }
    __syncthreads();
    if (tid == 0) {
        float s2 = 0.f;
        for (int k = 0; k < EXCH; ++k) s2 += pr_[k];
        pt_[0] = 1.0f / sqrtf(s2);
    }
    __syncthreads();
    {
        float rs = pt_[0];
        for (int i = tid; i < EXCH * EXCH; i += DCH) sWn[i] *= rs;
    }
    __syncthreads();

    // ---- loop-invariant W A-fragments (waves 0..1), hi/lo f16 split -------
    const int lane = tid & 31, hf = lane >> 4, l16 = lane & 15;
    v16h awhi, awlo;
    if (tid < 64) {
        const int w = tid >> 5;          // 16-row tile of W
#pragma unroll
        for (int v = 0; v < 8; ++v) {
            int k0  = (v >> 2) * 16 + hf * 8 + (v & 3) * 2;
            int row = 16 * w + l16;
            float x0 = sWn[row * EXCH + k0];
            float x1 = sWn[row * EXCH + k0 + 1];
            _Float16 h0 = (_Float16)x0, h1 = (_Float16)x1;
            awhi[2 * v]     = h0;  awhi[2 * v + 1] = h1;
            awlo[2 * v]     = (_Float16)(x0 - (float)h0);
            awlo[2 * v + 1] = (_Float16)(x1 - (float)h1);
        }
    }

    // ---- 8 microbatches, state h in a register the whole time -------------
    float h = 0.0f;
    for (int mb = 0; mb < NMB; ++mb) {
        const int phase = mb & 3;
        const int tb    = mb * MBSZ;

        // 256-step recurrence h = a*h + b (gathers hot in L2, prefetch ahead)
        if (a_tab) {
            for (int t0 = 0; t0 < MBSZ; t0 += 4) {
                if (t0 + 4 < MBSZ) {
                    int tn = toks[tb + t0 + 4];
                    __builtin_prefetch(&a_tab[(size_t)tn * DCH + ch], 0, 0);
                    __builtin_prefetch(&b_tab[(size_t)tn * DCH + ch], 0, 0);
                }
#pragma unroll
                for (int j = 0; j < 4; ++j) {
                    int tok = toks[tb + t0 + j];
                    h = fmaf(a_tab[(size_t)tok * DCH + ch], h,
                             b_tab[(size_t)tok * DCH + ch]);
                }
            }
        } else {
#pragma unroll 4
            for (int t = 0; t < MBSZ; ++t) {
                int   tok = toks[tb + t];
                float x   = a_raw[(size_t)tok * DCH + ch];
                float av  = AMIN + ARANGE / (1.0f + expf(-x));
                h = fmaf(av, h, b_tab[(size_t)tok * DCH + ch]);
            }
        }

        // Hadamard-64: stages 1..16 in-wave via shuffles, stage 32 via LDS
#pragma unroll
        for (int step = 1; step <= 16; step <<= 1) {
            float pv = __shfl_xor(h, step, 32);
            h = (ch & step) ? (pv - h) : (h + pv);
        }
        sl[ch] = h;
        __syncthreads();
        { float pv = sl[ch ^ 32]; h = (ch & 32) ? (pv - h) : (h + pv); }
        __syncthreads();

        h = tanhf(h);                    // T_ACT == 1.0
        sl[ch] = h;
        __syncthreads();

        // u[g*16+p] = sum_j local[g, (4p+j+phase) mod 256]
        if (ch < EXCH) {
            int p = ch & 15;
            float u = 0.f;
#pragma unroll
            for (int j = 0; j < 4; ++j)
                u += sl[(ch >> 4) * 256 + ((4 * p + j + phase) & 255)];
            su[ch] = u;
        }
        __syncthreads();

        // u_out = W @ u via WMMA: B has u in column 0 only; 3 WMMAs hi/lo
        if (tid < 64) {
            const int w = tid >> 5;
            v16h buh, bul;
#pragma unroll
            for (int v = 0; v < 8; ++v) {
                int k0 = (v >> 2) * 16 + hf * 8 + (v & 3) * 2;
                float x0 = (l16 == 0) ? su[k0]     : 0.f;
                float x1 = (l16 == 0) ? su[k0 + 1] : 0.f;
                _Float16 h0 = (_Float16)x0, h1 = (_Float16)x1;
                buh[2 * v]     = h0;  buh[2 * v + 1] = h1;
                bul[2 * v]     = (_Float16)(x0 - (float)h0);
                bul[2 * v + 1] = (_Float16)(x1 - (float)h1);
            }
            v8f acc = {};
            acc = __builtin_amdgcn_wmma_f32_16x16x32_f16(false, awlo, false, buh,
                                                         (short)0, acc, false, false);
            acc = __builtin_amdgcn_wmma_f32_16x16x32_f16(false, awhi, false, bul,
                                                         (short)0, acc, false, false);
            acc = __builtin_amdgcn_wmma_f32_16x16x32_f16(false, awhi, false, buh,
                                                         (short)0, acc, false, false);
            if (l16 == 0) {              // column 0 of D = W @ u
#pragma unroll
                for (int r = 0; r < 8; ++r)
                    suo[16 * w + 8 * hf + r] = acc[r];
            }
        }
        __syncthreads();

        // injection: local[g, (4p+j+phase)%256] += u_out[g*16+p] * 2^-3
        {
            int pos = ch & 255;
            int q   = (pos - phase) & 255;
            if (q < 64) h += suo[(ch >> 8) * 16 + (q >> 2)] * INJSC;
        }
        __syncthreads();                 // sl/su/suo safe to reuse next mb
    }

    out[batch * DCH + ch] = h;
}

// ---------------------------------------------------------------------------
// Host launcher: 2 launches total (a_tab precompute + fused kernel).
// ---------------------------------------------------------------------------
extern "C" void kernel_launch(void* const* d_in, const int* in_sizes, int n_in,
                              void* d_out, int out_size, void* d_ws, size_t ws_size,
                              hipStream_t stream) {
    const int*   tokens = (const int*)d_in[0];    // 32x2048
    const float* a_raw  = (const float*)d_in[1];  // 256x512
    const float* b_tab  = (const float*)d_in[2];  // 256x512
    const float* exw    = (const float*)d_in[3];  // 32x32

    float* atab = (ws_size >= 131072 * sizeof(float)) ? (float*)d_ws : nullptr;
    if (atab) atab_kernel<<<512, 256, 0, stream>>>(a_raw, atab);

    monoid_fused_kernel<<<BSZ, DCH, 0, stream>>>(tokens, a_raw, b_tab, atab,
                                                 exw, (float*)d_out);
}
